// CrossAttention_85023172592375
// MI455X (gfx1250) — matrix operand order repaired and checked by reading, compiled, not run on hardware
//
#include <hip/hip_runtime.h>
#include <hip/hip_bf16.h>

typedef __attribute__((ext_vector_type(16))) __bf16 v16bf;
typedef __attribute__((ext_vector_type(8)))  float  v8f;
typedef __attribute__((ext_vector_type(4)))  unsigned int v4u;
typedef __attribute__((ext_vector_type(8)))  int    v8i;
typedef __attribute__((ext_vector_type(4)))  int    v4i;

#if defined(__has_builtin)
#if __has_builtin(__builtin_amdgcn_tensor_load_to_lds) && __has_builtin(__builtin_amdgcn_s_wait_tensorcnt)
#define HAVE_TDM 1
#endif
#endif
#ifndef HAVE_TDM
#define HAVE_TDM 0
#endif

union FragAB { v16bf v; unsigned int u[8]; };

__device__ __forceinline__ unsigned short f32_to_bf16(float f) {
  union { float f; unsigned int u; } x; x.f = f;
  unsigned int r = x.u + 0x7FFFu + ((x.u >> 16) & 1u);   // RNE
  return (unsigned short)(r >> 16);
}

__device__ __forceinline__ v8f wmma_bf16(const FragAB& a, const FragAB& b, v8f c) {
  return __builtin_amdgcn_wmma_f32_16x16x32_bf16(
      /*neg_a=*/false, a.v, /*neg_b=*/false, b.v,
      /*c_mod=*/(short)0, c, /*reuse_a=*/false, /*reuse_b=*/false);
}

// ---- Async global->LDS 16B copy (GLOBAL_LOAD_ASYNC_TO_LDS_B128, ASYNCcnt) ----
__device__ __forceinline__ void async_copy_b128(const void* gptr, void* lds_generic) {
  unsigned lds_off = (unsigned)(unsigned long long)lds_generic;  // low 32b of generic LDS addr
  asm volatile("global_load_async_to_lds_b128 %0, %1, off"
               :: "v"(lds_off), "v"((unsigned long long)gptr) : "memory");
}
__device__ __forceinline__ void async_wait0() {
  asm volatile("s_wait_asynccnt 0" ::: "memory");
}

#if HAVE_TDM
// ---- TDM: DMA a 64x64 tile of 2-byte elements (row stride in elements) to LDS
// D# per ISA 8.3/8.4: group0 = {count, lds_addr, global_addr[56:0], type=2},
// group1 = {data_size=1(2B), tensor_dim0/1, tile_dim0/1=64, dim0_stride}.
// Toolchain uses the 6-arg builtin: (v4u, v8i, v4i, v4i, v8i, i32 cpol).
__device__ __forceinline__ void tdm_load_tile64x64_b16(
    const void* gptr, void* lds_generic,
    unsigned tensor_d0, unsigned tensor_d1, unsigned stride_elems)
{
  unsigned long long ga = (unsigned long long)gptr;
  unsigned lds = (unsigned)(unsigned long long)lds_generic;
  v4u g0;
  g0.x = 1u;                                                // count=1, user D#
  g0.y = lds;                                               // lds_addr[31:0]
  g0.z = (unsigned)(ga & 0xffffffffu);                      // global_addr[31:0]
  g0.w = (unsigned)((ga >> 32) & 0x01ffffffu) | (2u << 30); // addr[56:32] | type=2
  v8i g1;
  g1[0] = (int)(1u << 16);                                  // data_size=1 -> 2 bytes
  g1[1] = (int)((tensor_d0 & 0xffffu) << 16);               // tensor_dim0[15:0] @ bits63:48
  g1[2] = (int)((tensor_d0 >> 16) | ((tensor_d1 & 0xffffu) << 16)); // dim0 hi | dim1 lo
  g1[3] = (int)((tensor_d1 >> 16) | (64u << 16));           // dim1 hi | tile_dim0=64
  g1[4] = 64;                                               // tile_dim1=64, tile_dim2=0
  g1[5] = (int)stride_elems;                                // tensor_dim0_stride[31:0]
  g1[6] = 0;                                                // stride hi | dim1_stride lo
  g1[7] = 0;
  v4i z4 = {0, 0, 0, 0};                                    // groups 2/3 unused (2D)
  v8i z8 = {0, 0, 0, 0, 0, 0, 0, 0};
  __builtin_amdgcn_tensor_load_to_lds(g0, g1, z4, z4, z8, 0);
}
#endif

// ---------------------------------------------------------------------------
// Tiled GEMM: C[M,N] = A[M,K] * B[K,N] (+bias). A f32 or bf16 row-major;
// B f32 row-major (converted to bf16 while staging into LDS); C bf16 or f32.
// Block 128x128, BK=32, 256 threads = 8 waves (4x2), wave = 32x64 via 2x4
// v_wmma_f32_16x16x32_bf16 accumulators. bf16-A staging uses async-to-LDS.
// ---------------------------------------------------------------------------
template<bool A_F32, bool OUT_F32>
__global__ __launch_bounds__(256) void gemm_wmma_k(
    const void* __restrict__ Aptr, long lda,
    const float* __restrict__ Bptr, long ldb,
    void* __restrict__ Cptr, long ldc,
    const float* __restrict__ bias, int K)
{
  __shared__ unsigned short Alds[128 * 32];       // row-major [m][k]
  __shared__ unsigned short Blds[16 * 128 * 2];   // pair-major [k/2][n][k%2]

  const int t    = threadIdx.x;
  const long m0  = (long)blockIdx.y * 128;
  const long n0  = (long)blockIdx.x * 128;
  const int wave = t >> 5, lane = t & 31, half = lane >> 4, ml = lane & 15;
  const int wm   = wave >> 1, wn = wave & 1;       // 4 (M) x 2 (N) wave grid

  v8f acc[2][4] = {};

  const int ar = t >> 1, ac = (t & 1) * 16;        // A staging: 2 thr / row
  const int bk = t >> 3, bc = (t & 7) * 16;        // B staging: 8 thr / row

  for (int kk = 0; kk < K; kk += 32) {
    // ---- stage A tile ----
    if constexpr (A_F32) {
      const float* Af = (const float*)Aptr + (m0 + ar) * lda + kk + ac;
      if (kk + 32 < K) __builtin_prefetch(Af + 32, 0, 0);   // global_prefetch_b8
      unsigned short tmp[16];
      #pragma unroll
      for (int i = 0; i < 16; i += 4) {
        float4 f = *(const float4*)(Af + i);
        tmp[i+0] = f32_to_bf16(f.x); tmp[i+1] = f32_to_bf16(f.y);
        tmp[i+2] = f32_to_bf16(f.z); tmp[i+3] = f32_to_bf16(f.w);
      }
      ((uint4*)&Alds[ar*32 + ac])[0] = ((const uint4*)tmp)[0];
      ((uint4*)&Alds[ar*32 + ac])[1] = ((const uint4*)tmp)[1];
    } else {
      // bf16 A: pure copy -> async global->LDS DMA, no VGPR round-trip
      const unsigned short* Ab = (const unsigned short*)Aptr + (m0 + ar) * lda + kk + ac;
      if (kk + 32 < K) __builtin_prefetch(Ab + 32, 0, 0);
      async_copy_b128(Ab,     &Alds[ar*32 + ac]);
      async_copy_b128(Ab + 8, &Alds[ar*32 + ac + 8]);
    }
    // ---- stage B tile pair-major so fragment dwords are one ds_load_b32 ----
    {
      const float* Bf = Bptr + (long)(kk + bk) * ldb + n0 + bc;
      if (kk + 32 < K) __builtin_prefetch(Bf + 32 * ldb, 0, 0);
      unsigned short* dst = &Blds[(bk >> 1) * 256 + (bk & 1)];
      #pragma unroll
      for (int i = 0; i < 16; i += 4) {
        float4 f = *(const float4*)(Bf + i);
        dst[(bc + i + 0) * 2] = f32_to_bf16(f.x);
        dst[(bc + i + 1) * 2] = f32_to_bf16(f.y);
        dst[(bc + i + 2) * 2] = f32_to_bf16(f.z);
        dst[(bc + i + 3) * 2] = f32_to_bf16(f.w);
      }
    }
    if constexpr (!A_F32) async_wait0();
    __syncthreads();

    // ---- fragments: contiguous k = 16*half + s, same order for A(16x32) and B(32x16) ----
    FragAB afr[2], bfr[4];
    #pragma unroll
    for (int i = 0; i < 2; i++) {
      const unsigned short* ab = &Alds[(wm*32 + i*16 + ml) * 32];
      #pragma unroll
      for (int v = 0; v < 8; v++) {
        int kb = 2*v + 16*half;   // A: K pairs per VGPR
        afr[i].u[v] = *(const unsigned int*)(ab + kb);
      }
    }
    const unsigned int* B32 = (const unsigned int*)Blds;
    #pragma unroll
    for (int j = 0; j < 4; j++) {
      int n = wn*64 + j*16 + ml;
      #pragma unroll
      for (int v = 0; v < 8; v++)
        bfr[j].u[v] = B32[(v + 8*half) * 128 + n];   // B: K=2v(+16*half)
    }
    #pragma unroll
    for (int i = 0; i < 2; i++)
      #pragma unroll
      for (int j = 0; j < 4; j++)
        acc[i][j] = wmma_bf16(afr[i], bfr[j], acc[i][j]);

    __syncthreads();
  }

  // ---- epilogue (C/D layout: VGPR r -> M = r + 8*half, N = lane&15) ----
  #pragma unroll
  for (int i = 0; i < 2; i++)
    for (int j = 0; j < 4; j++)
      #pragma unroll
      for (int r = 0; r < 8; r++) {
        long m = m0 + wm*32 + i*16 + r + 8*half;
        long n = n0 + wn*64 + j*16 + ml;
        float val = acc[i][j][r];
        if constexpr (OUT_F32) {
          ((float*)Cptr)[m * ldc + n] = val + bias[n];
        } else {
          ((unsigned short*)Cptr)[m * ldc + n] = f32_to_bf16(val);
        }
      }
}

// ---------------------------------------------------------------------------
// Flash attention: one workgroup per (b, h, 64-row q tile); 4 waves, each
// owning 16 q rows. Q/K tiles DMA'd by TDM (tensor_load_to_lds, TENSORcnt);
// V tile transposed manually into [hd][kv] so P.V B-frag pairs are contiguous.
// Online softmax with running (m, l) per row.
// KVp layout: [16384][2048] bf16, cols 0..1023 = K heads, 1024..2047 = V heads.
// ---------------------------------------------------------------------------
__global__ __launch_bounds__(128) void attn_k(
    const unsigned short* __restrict__ Qp,   // [4096][1024] bf16 (B,Nq,H,hd)
    const unsigned short* __restrict__ KVp,  // [16384][2048] bf16
    unsigned short* __restrict__ Op)         // [4096][1024] bf16
{
  __shared__ unsigned short Qs[64 * 64];
  __shared__ unsigned short Ks[64 * 64];
  __shared__ unsigned short Vt[64 * 64];       // transposed [hd][kv]
  __shared__ unsigned short Ps[4][16 * 64];    // per-wave P tile

  const int t = threadIdx.x, wave = t >> 5, lane = t & 31;
  const int half = lane >> 4, ml = lane & 15;
  const int qt = blockIdx.x, h = blockIdx.y, b = blockIdx.z;
  const float scale = 0.125f;                  // 64^-0.5

  // ---- load Q tile once ----
#if HAVE_TDM
  if (wave == 0) {
    tdm_load_tile64x64_b16(Qp + (long)(b*1024 + qt*64) * 1024 + h*64, Qs,
                           1024u, 4096u, 1024u);
    __builtin_amdgcn_s_wait_tensorcnt(0);
  }
#else
  {
    int r = t >> 1, c0 = (t & 1) * 32;
    const uint4* src = (const uint4*)(Qp + ((long)(b*1024 + qt*64 + r)) * 1024 + h*64 + c0);
    uint4* dst = (uint4*)&Qs[r*64 + c0];
    dst[0] = src[0]; dst[1] = src[1]; dst[2] = src[2]; dst[3] = src[3];
  }
#endif

  float mi[8], li[8];
  #pragma unroll
  for (int r = 0; r < 8; r++) { mi[r] = -1e30f; li[r] = 0.f; }
  v8f o[4] = {};

  for (int kt = 0; kt < 4096; kt += 64) {
    __syncthreads();
    const long kvrow0 = (long)(b*4096 + kt);
    // ---- K tile: TDM DMA (row stride 2048 elems), overlaps V staging ----
#if HAVE_TDM
    if (wave == 0)
      tdm_load_tile64x64_b16(KVp + kvrow0 * 2048 + h*64, Ks, 2048u, 16384u, 2048u);
#else
    {
      int r = t >> 1, c0 = (t & 1) * 32;
      const unsigned short* krow = KVp + (kvrow0 + r) * 2048 + h*64 + c0;
      const uint4* ks4 = (const uint4*)krow;
      uint4* kd = (uint4*)&Ks[r*64 + c0];
      kd[0] = ks4[0]; kd[1] = ks4[1]; kd[2] = ks4[2]; kd[3] = ks4[3];
    }
#endif
    { // ---- V tile transposed (manual: DMA cannot transpose) ----
      int r = t >> 1, c0 = (t & 1) * 32;
      const unsigned short* vrow = KVp + (kvrow0 + r) * 2048 + 1024 + h*64 + c0;
      #pragma unroll
      for (int jj = 0; jj < 32; jj++)
        Vt[(c0 + jj)*64 + r] = vrow[jj];
    }
#if HAVE_TDM
    if (wave == 0) __builtin_amdgcn_s_wait_tensorcnt(0);
#endif
    __syncthreads();

    // ---- S = Q K^T : 16(q) x 64(kv), K-dim = hd = 64 -> 2 k-steps ----
    v8f s[4] = {};
    #pragma unroll
    for (int ks = 0; ks < 2; ks++) {
      FragAB a;
      const unsigned short* ab = &Qs[(wave*16 + ml)*64 + ks*32];
      #pragma unroll
      for (int v = 0; v < 8; v++) {
        int kb = 2*v + 16*half;
        a.u[v] = *(const unsigned int*)(ab + kb);
      }
      #pragma unroll
      for (int j = 0; j < 4; j++) {
        FragAB bf;
        #pragma unroll
        for (int v = 0; v < 8; v++) {
          int kb2 = 2*v + 16*half + ks*32;
          bf.u[v] = *(const unsigned int*)&Ks[(j*16 + ml)*64 + kb2];
        }
        s[j] = wmma_bf16(a, bf, s[j]);
      }
    }

    // ---- online softmax: rows live on 16-lane halves -> shfl_xor 1/2/4/8 ----
    float mnew[8], alpha[8];
    #pragma unroll
    for (int r = 0; r < 8; r++) {
      float v0 = fmaxf(fmaxf(s[0][r], s[1][r]), fmaxf(s[2][r], s[3][r])) * scale;
      v0 = fmaxf(v0, __shfl_xor(v0, 1, 32));
      v0 = fmaxf(v0, __shfl_xor(v0, 2, 32));
      v0 = fmaxf(v0, __shfl_xor(v0, 4, 32));
      v0 = fmaxf(v0, __shfl_xor(v0, 8, 32));
      mnew[r] = fmaxf(mi[r], v0);
      alpha[r] = __expf(mi[r] - mnew[r]);
      mi[r] = mnew[r];
    }
    #pragma unroll
    for (int j = 0; j < 4; j++)
      #pragma unroll
      for (int r = 0; r < 8; r++)
        s[j][r] = __expf(s[j][r] * scale - mnew[r]);
    #pragma unroll
    for (int r = 0; r < 8; r++) {
      float sm = s[0][r] + s[1][r] + s[2][r] + s[3][r];
      sm += __shfl_xor(sm, 1, 32);
      sm += __shfl_xor(sm, 2, 32);
      sm += __shfl_xor(sm, 4, 32);
      sm += __shfl_xor(sm, 8, 32);
      li[r] = li[r] * alpha[r] + sm;
      #pragma unroll
      for (int j = 0; j < 4; j++) o[j][r] *= alpha[r];
    }

    // ---- restage P (C-layout) as bf16 A-operand in per-wave LDS ----
    unsigned short* pw = Ps[wave];
    #pragma unroll
    for (int j = 0; j < 4; j++)
      #pragma unroll
      for (int r = 0; r < 8; r++)
        pw[(r + 8*half)*64 + j*16 + ml] = f32_to_bf16(s[j][r]);

    // ---- O += P V : K-dim = kv = 64 -> 2 k-steps (in-wave LDS, in-order) ----
    #pragma unroll
    for (int ks = 0; ks < 2; ks++) {
      FragAB a;
      const unsigned short* ab = &pw[ml*64 + ks*32];
      #pragma unroll
      for (int v = 0; v < 8; v++) {
        int kb = 2*v + 16*half;
        a.u[v] = *(const unsigned int*)(ab + kb);
      }
      #pragma unroll
      for (int j = 0; j < 4; j++) {
        FragAB bf;
        #pragma unroll
        for (int v = 0; v < 8; v++) {
          int kb2 = 2*v + 16*half + ks*32;
          bf.u[v] = *(const unsigned int*)&Vt[(j*16 + ml)*64 + kb2];
        }
        o[j] = wmma_bf16(a, bf, o[j]);
      }
    }
  }

  // ---- normalize and emit O as bf16 [B,Nq,H,hd] ----
  #pragma unroll
  for (int j = 0; j < 4; j++)
    #pragma unroll
    for (int r = 0; r < 8; r++) {
      long row = b*1024 + qt*64 + wave*16 + r + 8*half;
      float val = o[j][r] / li[r];
      Op[row * 1024 + h*64 + j*16 + ml] = f32_to_bf16(val);
    }
}

// ---------------------------------------------------------------------------
extern "C" void kernel_launch(void* const* d_in, const int* in_sizes, int n_in,
                              void* d_out, int out_size, void* d_ws, size_t ws_size,
                              hipStream_t stream) {
  (void)in_sizes; (void)n_in; (void)out_size; (void)ws_size;
  const float* q     = (const float*)d_in[0];   // [4,1024,1024]
  const float* kv    = (const float*)d_in[1];   // [4,4096,1024]
  const float* Wqkv  = (const float*)d_in[2];   // [1024,3072]
  const float* Wproj = (const float*)d_in[3];   // [1024,1024]
  const float* bproj = (const float*)d_in[4];   // [1024]
  float* out = (float*)d_out;                   // [4,1024,1024] f32

  char* ws = (char*)d_ws;
  unsigned short* KVp = (unsigned short*)ws;                               // 64 MB
  unsigned short* Qp  = (unsigned short*)(ws + (size_t)16384*2048*2);      //  8 MB
  unsigned short* Op  = (unsigned short*)(ws + (size_t)16384*2048*2
                                             + (size_t)4096*1024*2);       //  8 MB

  dim3 blk(256);
  // 1) K/V projection: kv[16384x1024] x Wqkv[:,1024:3072] -> bf16 [16384x2048]
  gemm_wmma_k<true, false><<<dim3(2048/128, 16384/128), blk, 0, stream>>>(
      kv, 1024, Wqkv + 1024, 3072, KVp, 2048, nullptr, 1024);
  // 2) Q projection: q[4096x1024] x Wqkv[:,0:1024] -> bf16 [4096x1024]
  gemm_wmma_k<true, false><<<dim3(1024/128, 4096/128), blk, 0, stream>>>(
      q, 1024, Wqkv, 3072, Qp, 1024, nullptr, 1024);
  // 3) attention: grid (q-tiles=16, heads=16, batch=4)
  attn_k<<<dim3(16, 16, 4), dim3(128), 0, stream>>>(Qp, KVp, Op);
  // 4) output projection + bias -> f32
  gemm_wmma_k<false, true><<<dim3(1024/128, 4096/128), blk, 0, stream>>>(
      Op, 1024, Wproj, 1024, out, 1024, bproj, 1024);
}
